// VisionGPTModel_7258494730872
// MI455X (gfx1250) — matrix operand order repaired and checked by reading, not compiled, not run on hardware
//
#include <hip/hip_runtime.h>

// ---------------------------------------------------------------------------
// GPT-style transformer forward for MI455X (gfx1250, wave32).
// Dense GEMMs: v_wmma_f32_16x16x32_bf16, block tile 128x64 (8 waves),
// wave tile 16x64 (4 independent accumulators sharing one A fragment).
// B K-slices are staged into LDS with the CDNA5 async DMA path
// (global_load_async_to_lds_b128 + s_wait_asynccnt), double-buffered so the
// next tile's staging overlaps the current tile's WMMAs.
// Attention is a streaming-softmax VALU kernel (~9% of FLOPs).
// ---------------------------------------------------------------------------

#define DD   384
#define HH   6
#define HSZ  64
#define LLAY 6
#define TTOK 1024
#define BSZ  8
#define MM   (BSZ * TTOK)   // 8192 token rows
#define DFF  (4 * DD)       // 1536
#define VV   32000

#define BSTRIDE 40          // padded LDS row stride (elements) for 64x32 B tile

typedef __attribute__((ext_vector_type(16))) __bf16 v16bf;
typedef __attribute__((ext_vector_type(8)))  float  v8f;

union FragBF {
  uint4 u[2];
  v16bf v;
};

// Async global->LDS 16B copy (CDNA5, ASYNCcnt). VDST is a VGPR holding the
// workgroup-relative LDS byte address (== low 32 bits of the flat shared
// pointer per the ISA aperture rules); VADDR is the 64-bit global address.
__device__ __forceinline__ void async_load_b128(unsigned lds_off, const void* gaddr) {
  asm volatile("global_load_async_to_lds_b128 %0, %1, off"
               :: "v"(lds_off), "v"(gaddr)
               : "memory");
}

// ---------------------------------------------------------------------------
// Embedding: x[m, d] = tok_emb[clip(idx[m])][d] + pos_emb[t][d]
// ---------------------------------------------------------------------------
__global__ void embed_kernel(const int* __restrict__ idx,
                             const float* __restrict__ tok,
                             const float* __restrict__ pos,
                             float* __restrict__ x) {
  int i = blockIdx.x * 256 + threadIdx.x;       // over MM*DD
  int d = i % DD;
  int m = i / DD;
  int t = m % TTOK;
  int id = idx[m];
  id = id < 0 ? 0 : (id > (VV - 1) ? (VV - 1) : id);
  x[i] = tok[(size_t)id * DD + d] + pos[(size_t)t * DD + d];
}

// ---------------------------------------------------------------------------
// Weight transpose + cast: src fp32 [Hn][K][N] -> dst bf16 [Hn*N][K]
// ---------------------------------------------------------------------------
__global__ void transpose_cast_kernel(const float* __restrict__ src,
                                      __bf16* __restrict__ dst,
                                      int K, int N, long total) {
  long i = (long)blockIdx.x * 256 + threadIdx.x;   // dst flat index
  if (i >= total) return;
  int  k = (int)(i % K);
  long r = i / K;
  int  n = (int)(r % N);
  int  h = (int)(r / N);
  dst[i] = (__bf16)src[((long)h * K + k) * N + n];
}

// ---------------------------------------------------------------------------
// fp32 -> bf16 elementwise cast
// ---------------------------------------------------------------------------
__global__ void cast_bf16_kernel(const float* __restrict__ src,
                                 __bf16* __restrict__ dst, long n) {
  long i = (long)blockIdx.x * 256 + threadIdx.x;
  if (i < n) dst[i] = (__bf16)src[i];
}

// ---------------------------------------------------------------------------
// LayerNorm over D=384: one wave32 per row (12 elems/lane), bf16 output.
// ---------------------------------------------------------------------------
__global__ void ln_kernel(const float* __restrict__ x,
                          const float* __restrict__ g,
                          const float* __restrict__ b,
                          __bf16* __restrict__ out) {
  int lane = threadIdx.x & 31;
  int row  = blockIdx.x * 8 + (threadIdx.x >> 5);
  const float* xr = x + (size_t)row * DD;
  float vals[12];
  float s = 0.f, ss = 0.f;
#pragma unroll
  for (int i = 0; i < 12; ++i) {
    float v = xr[lane + 32 * i];
    vals[i] = v;
    s += v;
    ss += v * v;
  }
#pragma unroll
  for (int m = 16; m >= 1; m >>= 1) {
    s  += __shfl_xor(s,  m, 32);
    ss += __shfl_xor(ss, m, 32);
  }
  float mu  = s * (1.f / DD);
  float var = ss * (1.f / DD) - mu * mu;
  float inv = rsqrtf(var + 1e-5f);
#pragma unroll
  for (int i = 0; i < 12; ++i) {
    int c = lane + 32 * i;
    out[(size_t)row * DD + c] = (__bf16)((vals[i] - mu) * inv * g[c] + b[c]);
  }
}

// ---------------------------------------------------------------------------
// WMMA GEMM: C[M,N] = A_bf16[M,K] * Bt_bf16[N,K]^T  (+bias, +residual, relu)
//
// Block: 256 threads = 8 waves; block tile 128(M) x 64(N).
// Wave w owns rows [m0+16w, m0+16w+16), all 64 block columns:
//   4 independent v8f accumulators reuse one A fragment per K-step.
// The shared 64x32 B slice is staged into LDS via async DMA
// (global_load_async_to_lds_b128, ASYNCcnt), double-buffered: the load for
// step s+1 is in flight while step s computes (wait asynccnt<=1).
//
// Fragment layouts (ISA 7.12.2/7.12.4):
//   A (16x32 bf16): lane=16h+m, elems 0..7 -> K=8h.., elems 8..15 -> K=16+8h..
//   B (32x16 bf16): lane=16h+n, elems 0..7 -> K=8h.., elems 8..15 -> K=16+8h..
//   C/D:            VGPR r -> M=8h+r, lane%16 -> N
// Requires M%128==0, N%64==0, K%32==0 (true for all GEMMs here).
// ---------------------------------------------------------------------------
__global__ void gemm_bf16_wmma_kernel(const __bf16* __restrict__ A,
                                      const __bf16* __restrict__ Bt,
                                      int M, int N, int K,
                                      const float* bias,   // [N] or null
                                      const float* res,    // [M,N] or null (may alias outF)
                                      float* outF,         // [M,N] or null
                                      __bf16* outB,        // [M,N] or null
                                      int relu) {
  __shared__ __bf16 Bs[2][64 * BSTRIDE];

  const int tid  = threadIdx.x;
  const int lane = tid & 31;
  const int wave = tid >> 5;
  const int hf   = lane >> 4;        // 0/1 half of wave
  const int l16  = lane & 15;

  const int m0 = blockIdx.y * 128 + wave * 16;
  const int n0 = blockIdx.x * 64;

  // B staging assignment: thread -> (row 0..63, 16B segment 0..3)
  const int brow = tid >> 2;
  const int bseg = tid & 3;
  const __bf16* Bg = Bt + (size_t)(n0 + brow) * K + bseg * 8;
  const unsigned ldsElem = (unsigned)(brow * BSTRIDE + bseg * 8);
  const unsigned lds0 = (unsigned)(uintptr_t)(&Bs[0][ldsElem]);
  const unsigned lds1 = (unsigned)(uintptr_t)(&Bs[1][ldsElem]);

  const __bf16* Ar = A + (size_t)(m0 + l16) * K + 8 * hf;

  v8f acc[4];
#pragma unroll
  for (int nt = 0; nt < 4; ++nt)
    acc[nt] = (v8f){0.f, 0.f, 0.f, 0.f, 0.f, 0.f, 0.f, 0.f};

  const int nsteps = K >> 5;
  // prologue: stage K-step 0 into buffer 0
  async_load_b128(lds0, Bg);

  for (int step = 0; step < nsteps; ++step) {
    const int cur = step & 1;
    if (step + 1 < nsteps) {
      // stage next K-slice into the other buffer; overlap with this step's WMMAs
      async_load_b128(cur ? lds0 : lds1, Bg + (size_t)(step + 1) * 32);
      asm volatile("s_wait_asynccnt 1" ::: "memory");   // current buffer ready
    } else {
      asm volatile("s_wait_asynccnt 0" ::: "memory");
    }
    __syncthreads();                                    // visible to all waves

    const int k = step * 32;
    FragBF fa;
    fa.u[0] = *reinterpret_cast<const uint4*>(Ar + k);        // K = 8h ..
    fa.u[1] = *reinterpret_cast<const uint4*>(Ar + k + 16);   // K = 16+8h ..
    if (k + 128 < K)
      __builtin_prefetch((const void*)(Ar + k + 128), 0, 3);  // global_prefetch (near)

#pragma unroll
    for (int nt = 0; nt < 4; ++nt) {
      FragBF fb;
      const __bf16* bsrc = &Bs[cur][(nt * 16 + l16) * BSTRIDE + 8 * hf];
      fb.u[0] = *reinterpret_cast<const uint4*>(bsrc);        // K = 8h ..
      fb.u[1] = *reinterpret_cast<const uint4*>(bsrc + 16);   // K = 16+8h ..
      acc[nt] = __builtin_amdgcn_wmma_f32_16x16x32_bf16(
          /*neg_a=*/false, fa.v, /*neg_b=*/false, fb.v,
          /*c_mod=*/(short)0, acc[nt], /*reuse_a=*/false, /*reuse_b=*/false);
    }
    __syncthreads();   // all waves done reading buf[cur] before it is refilled
  }

  // epilogue
#pragma unroll
  for (int nt = 0; nt < 4; ++nt) {
    const int n = n0 + nt * 16 + l16;
    const float bn = bias ? bias[n] : 0.f;
#pragma unroll
    for (int r = 0; r < 8; ++r) {
      const int    mrow = m0 + 8 * hf + r;
      const size_t off  = (size_t)mrow * N + n;
      float v = acc[nt][r] + bn;
      if (res)  v += res[off];
      if (relu) v = fmaxf(v, 0.f);
      if (outF) outF[off] = v;
      if (outB) outB[off] = (__bf16)v;
    }
  }
}

// ---------------------------------------------------------------------------
// Causal attention, streaming softmax. One wave32 per query row (b,h,t);
// each lane owns head-dims 2*lane, 2*lane+1 (HS=64). qkv layout per token
// row: [q(384) | k(384) | v(384)], per-head columns h*64+d.
// ---------------------------------------------------------------------------
__global__ void attn_kernel(const __bf16* __restrict__ qkv,
                            __bf16* __restrict__ out) {
  const int lane = threadIdx.x & 31;
  const int rid  = blockIdx.x * 8 + (threadIdx.x >> 5);   // 0..BSZ*HH*TTOK-1
  const int b  = rid / (HH * TTOK);
  const int h  = (rid / TTOK) % HH;
  const int t  = rid % TTOK;

  const __bf16* qrow = qkv + (size_t)(b * TTOK + t) * (3 * DD) + h * HSZ;
  const float q0 = (float)qrow[2 * lane];
  const float q1 = (float)qrow[2 * lane + 1];
  const float scale = 0.125f;   // HS^-0.5

  const __bf16* kbase = qkv + (size_t)b * TTOK * (3 * DD) + DD + h * HSZ;
  const __bf16* vbase = kbase + DD;

  float m = -INFINITY, ssum = 0.f, a0 = 0.f, a1 = 0.f;
  for (int s = 0; s <= t; ++s) {
    const __bf16* kr = kbase + (size_t)s * (3 * DD);
    float part = q0 * (float)kr[2 * lane] + q1 * (float)kr[2 * lane + 1];
#pragma unroll
    for (int w = 16; w >= 1; w >>= 1) part += __shfl_xor(part, w, 32);
    const float sc = part * scale;
    const float nm = fmaxf(m, sc);
    const float f  = __expf(m - nm);     // exp(-inf)=0 on first iteration
    const float p  = __expf(sc - nm);
    const __bf16* vr = vbase + (size_t)s * (3 * DD);
    a0   = a0 * f + p * (float)vr[2 * lane];
    a1   = a1 * f + p * (float)vr[2 * lane + 1];
    ssum = ssum * f + p;
    m = nm;
  }
  const float inv = 1.f / ssum;
  __bf16* orow = out + (size_t)(b * TTOK + t) * DD + h * HSZ;
  orow[2 * lane]     = (__bf16)(a0 * inv);
  orow[2 * lane + 1] = (__bf16)(a1 * inv);
}

// ---------------------------------------------------------------------------
// Host launch
// ---------------------------------------------------------------------------
extern "C" void kernel_launch(void* const* d_in, const int* in_sizes, int n_in,
                              void* d_out, int out_size, void* d_ws, size_t ws_size,
                              hipStream_t stream) {
  (void)in_sizes; (void)n_in; (void)out_size; (void)ws_size;

  const int*   inp  = (const int*)  d_in[0];
  const float* tok  = (const float*)d_in[1];
  const float* pos  = (const float*)d_in[2];
  const float* Wq   = (const float*)d_in[3];
  const float* Wk   = (const float*)d_in[4];
  const float* Wv   = (const float*)d_in[5];
  const float* Wo   = (const float*)d_in[6];
  const float* bo   = (const float*)d_in[7];
  const float* ln1g = (const float*)d_in[8];
  const float* ln1b = (const float*)d_in[9];
  const float* ln2g = (const float*)d_in[10];
  const float* ln2b = (const float*)d_in[11];
  const float* W1   = (const float*)d_in[12];
  const float* b1   = (const float*)d_in[13];
  const float* W2   = (const float*)d_in[14];
  const float* b2   = (const float*)d_in[15];
  const float* lmw  = (const float*)d_in[16];
  const float* lmb  = (const float*)d_in[17];

  // --- workspace carve-out (256B aligned blocks) ---
  char* wp = (char*)d_ws;
  auto carve = [&wp](size_t bytes) {
    char* p = wp;
    wp += (bytes + 255) & ~(size_t)255;
    return p;
  };
  __bf16* lmT   = (__bf16*)carve((size_t)VV * DD * 2);           // [V][D]
  __bf16* qkvT  = (__bf16*)carve((size_t)LLAY * 3 * DD * DD * 2);// per layer [1152][384]
  __bf16* woT   = (__bf16*)carve((size_t)LLAY * DD * DD * 2);    // per layer [384][384]
  __bf16* w1T   = (__bf16*)carve((size_t)LLAY * DFF * DD * 2);   // per layer [1536][384]
  __bf16* w2T   = (__bf16*)carve((size_t)LLAY * DD * DFF * 2);   // per layer [384][1536]
  float*  x     = (float*) carve((size_t)MM * DD * 4);
  __bf16* hbuf  = (__bf16*)carve((size_t)MM * DD * 2);
  __bf16* qkv   = (__bf16*)carve((size_t)MM * 3 * DD * 2);
  __bf16* attn  = (__bf16*)carve((size_t)MM * DD * 2);
  __bf16* ff    = (__bf16*)carve((size_t)MM * DFF * 2);

  auto nblk = [](size_t total) { return dim3((unsigned)((total + 255) / 256)); };

  // --- weight prep: transpose + bf16 cast ---
  transpose_cast_kernel<<<nblk((size_t)VV * DD), 256, 0, stream>>>(
      lmw, lmT, DD, VV, (long)VV * DD);
  for (int l = 0; l < LLAY; ++l) {
    const size_t ws = (size_t)HH * DD * HSZ;      // 147456 per layer
    __bf16* dst = qkvT + (size_t)l * 3 * DD * DD;
    transpose_cast_kernel<<<nblk(ws), 256, 0, stream>>>(Wq + l * ws, dst,
                                                        DD, HSZ, (long)ws);
    transpose_cast_kernel<<<nblk(ws), 256, 0, stream>>>(Wk + l * ws, dst + (size_t)DD * DD,
                                                        DD, HSZ, (long)ws);
    transpose_cast_kernel<<<nblk(ws), 256, 0, stream>>>(Wv + l * ws, dst + (size_t)2 * DD * DD,
                                                        DD, HSZ, (long)ws);
    transpose_cast_kernel<<<nblk((size_t)DD * DD), 256, 0, stream>>>(
        Wo + (size_t)l * DD * DD, woT + (size_t)l * DD * DD, DD, DD, (long)DD * DD);
    transpose_cast_kernel<<<nblk((size_t)DD * DFF), 256, 0, stream>>>(
        W1 + (size_t)l * DD * DFF, w1T + (size_t)l * DFF * DD, DD, DFF, (long)DD * DFF);
    transpose_cast_kernel<<<nblk((size_t)DFF * DD), 256, 0, stream>>>(
        W2 + (size_t)l * DFF * DD, w2T + (size_t)l * DD * DFF, DFF, DD, (long)DFF * DD);
  }

  // --- embedding ---
  embed_kernel<<<nblk((size_t)MM * DD), 256, 0, stream>>>(inp, tok, pos, x);

  // --- transformer layers ---
  for (int l = 0; l < LLAY; ++l) {
    ln_kernel<<<MM / 8, 256, 0, stream>>>(x, ln1g + l * DD, ln1b + l * DD, hbuf);

    gemm_bf16_wmma_kernel<<<dim3(3 * DD / 64, MM / 128), 256, 0, stream>>>(
        hbuf, qkvT + (size_t)l * 3 * DD * DD, MM, 3 * DD, DD,
        nullptr, nullptr, nullptr, qkv, 0);

    attn_kernel<<<(BSZ * HH * TTOK) / 8, 256, 0, stream>>>(qkv, attn);

    gemm_bf16_wmma_kernel<<<dim3(DD / 64, MM / 128), 256, 0, stream>>>(
        attn, woT + (size_t)l * DD * DD, MM, DD, DD,
        bo + l * DD, x, x, nullptr, 0);

    ln_kernel<<<MM / 8, 256, 0, stream>>>(x, ln2g + l * DD, ln2b + l * DD, hbuf);

    gemm_bf16_wmma_kernel<<<dim3(DFF / 64, MM / 128), 256, 0, stream>>>(
        hbuf, w1T + (size_t)l * DFF * DD, MM, DFF, DD,
        b1 + l * DFF, nullptr, nullptr, ff, 1);

    gemm_bf16_wmma_kernel<<<dim3(DD / 64, MM / 128), 256, 0, stream>>>(
        ff, w2T + (size_t)l * DD * DFF, MM, DD, DFF,
        b2 + l * DD, x, x, nullptr, 0);
  }

  // --- LM head: logits = x @ lm_w + lm_b ---
  cast_bf16_kernel<<<nblk((size_t)MM * DD), 256, 0, stream>>>(x, hbuf, (long)MM * DD);
  gemm_bf16_wmma_kernel<<<dim3(VV / 64, MM / 128), 256, 0, stream>>>(
      hbuf, lmT, MM, VV, DD,
      lmb, nullptr, (float*)d_out, nullptr, 0);
}
